// FFiNetModel_35862976922278
// MI455X (gfx1250) — matrix operand order repaired and checked by reading, not compiled, not run on hardware
//
#include <hip/hip_runtime.h>
#include <hip/hip_bf16.h>
#include <math.h>

// ---------------------------------------------------------------------------
// FFiNet forward for MI455X (gfx1250, wave32).
// Dense projections use V_WMMA_F32_16X16X32_BF16 (each wave owns a 16x64
// output strip: one A-fragment feeds 4 independent WMMAs; B addresses are
// pointer-increment + immediate-offset so the hot loop is loads+WMMA only);
// graph attention uses wave-per-item kernels with LDS-staged weights,
// shfl reductions, and atomics.
// ---------------------------------------------------------------------------

typedef __attribute__((ext_vector_type(16))) __bf16 v16bf;
typedef __attribute__((ext_vector_type(8)))  float  v8f;

union FragBF { v16bf v; uint4 q[2]; unsigned short s[16]; };

// ---- helpers ---------------------------------------------------------------

__device__ __forceinline__ unsigned short f2bf(float f) {
  unsigned u = __float_as_uint(f);
  unsigned r = u + 0x7FFFu + ((u >> 16) & 1u);   // round-to-nearest-even
  return (unsigned short)(r >> 16);
}

// order-preserving float<->uint for atomicMax on floats (handles negatives)
__device__ __forceinline__ unsigned int ford(float f) {
  unsigned u = __float_as_uint(f);
  return (u & 0x80000000u) ? ~u : (u | 0x80000000u);
}
__device__ __forceinline__ float ford_inv(unsigned int u) {
  unsigned o = (u & 0x80000000u) ? (u & 0x7FFFFFFFu) : ~u;
  return __uint_as_float(o);
}

__device__ __forceinline__ float distc(const float* pos, int a, int b) {
  float dx = pos[a*3+0]-pos[b*3+0];
  float dy = pos[a*3+1]-pos[b*3+1];
  float dz = pos[a*3+2]-pos[b*3+2];
  float d = sqrtf(dx*dx+dy*dy+dz*dz + 1e-12f);
  return fminf(fmaxf(d, 0.05f), 10.0f);
}

__device__ __forceinline__ float angle_at(const float* pos, int ia, int im, int ib) {
  float ux = pos[ia*3+0]-pos[im*3+0], uy = pos[ia*3+1]-pos[im*3+1], uz = pos[ia*3+2]-pos[im*3+2];
  float vx = pos[ib*3+0]-pos[im*3+0], vy = pos[ib*3+1]-pos[im*3+1], vz = pos[ib*3+2]-pos[im*3+2];
  float nu = sqrtf(ux*ux+uy*uy+uz*uz + 1e-12f);
  float nv = sqrtf(vx*vx+vy*vy+vz*vz + 1e-12f);
  float c = (ux*vx+uy*vy+uz*vz) / (nu*nv + 1e-12f);
  c = fminf(fmaxf(c, -1.0f + 1e-7f), 1.0f - 1e-7f);
  return acosf(c);
}

__device__ __forceinline__ float dihedral4(const float* pos, int i0, int i1, int i2, int i3) {
  float b1x = pos[i1*3+0]-pos[i0*3+0], b1y = pos[i1*3+1]-pos[i0*3+1], b1z = pos[i1*3+2]-pos[i0*3+2];
  float b2x = pos[i2*3+0]-pos[i1*3+0], b2y = pos[i2*3+1]-pos[i1*3+1], b2z = pos[i2*3+2]-pos[i1*3+2];
  float b3x = pos[i3*3+0]-pos[i2*3+0], b3y = pos[i3*3+1]-pos[i2*3+1], b3z = pos[i3*3+2]-pos[i2*3+2];
  float n1x = b1y*b2z - b1z*b2y, n1y = b1z*b2x - b1x*b2z, n1z = b1x*b2y - b1y*b2x;
  float n2x = b2y*b3z - b2z*b3y, n2y = b2z*b3x - b2x*b3z, n2z = b2x*b3y - b2y*b3x;
  float nb2 = sqrtf(b2x*b2x+b2y*b2y+b2z*b2z + 1e-12f) + 1e-12f;
  float ux = b2x/nb2, uy = b2y/nb2, uz = b2z/nb2;
  float m1x = n1y*uz - n1z*uy, m1y = n1z*ux - n1x*uz, m1z = n1x*uy - n1y*ux;
  float sy = m1x*n2x + m1y*n2y + m1z*n2z;
  float cx = n1x*n2x + n1y*n2y + n1z*n2z + 1e-12f;
  return atan2f(sy, cx);
}

// 40-center RBF dot products for 8 heads (channel c = jj*32 + lane).
// sW = 40x256 weight matrix in LDS. gate folds the edge_attr mask.
__device__ __forceinline__ void rbf_dot8(float dc, float gate, const float* sW,
                                         int lane, float rd[8]) {
  float t0 = dc - 0.25f * (float)lane;
  float ra = expf(-40.0f * t0 * t0) * gate;
  float rb = 0.0f;
  if (lane < 8) { float t1 = dc - 0.25f * (float)(lane + 32); rb = expf(-40.0f*t1*t1) * gate; }
  for (int j = 0; j < 40; ++j) {
    float rj = (j < 32) ? __shfl(ra, j) : __shfl(rb, j - 32);
    const float* w = sW + j*256 + lane;
#pragma unroll
    for (int jj = 0; jj < 8; ++jj) rd[jj] += rj * w[jj*32];
  }
}

__device__ __forceinline__ float wave_sum(float p) {
#pragma unroll
  for (int off = 16; off > 0; off >>= 1) p += __shfl_xor(p, off);
  return p;
}

// ---- conversion / packing --------------------------------------------------

__global__ void k_f32_to_bf16(const float* __restrict__ in,
                              unsigned short* __restrict__ out, long long n) {
  long long i = (long long)blockIdx.x * blockDim.x + threadIdx.x;
  if (i < n) out[i] = f2bf(in[i]);
}

// Repack W[K x 256] (f32 row-major) into the wmma_bf16 B-fragment layout:
// Bp[((ct*Kc + kc)*32 + lane)*16 + e] = bf16( W[(kc*32 + (lane>>4)*16 + e)*256 + ct*16 + (lane&15)] )
__global__ void k_pack_w(const float* __restrict__ W, unsigned short* __restrict__ Bp, int K) {
  int idx = blockIdx.x * blockDim.x + threadIdx.x;
  int total = K * 256;
  if (idx >= total) return;
  int e    = idx & 15;
  int lane = (idx >> 4) & 31;
  int rest = idx >> 9;
  int kcn  = K >> 5;
  int kc   = rest % kcn;
  int ct   = rest / kcn;
  int k    = kc*32 + (lane >> 4)*16 + e;
  int col  = ct*16 + (lane & 15);
  Bp[idx] = f2bf(W[(size_t)k*256 + col]);
}

// ---- bf16 WMMA GEMM: C[M x 256] (+)= A[M x K] @ Bp, + bias + sinusoidal PE --
// One wave per 16x64 output strip: one A-fragment load feeds 4 independent
// v_wmma_f32_16x16x32_bf16 ops. Column tiles and the +8-element A run sit at
// unroll-constant byte offsets (<24-bit IOFFSET), so the inner loop is just
// two pointer increments + b128 load clauses + 4 WMMAs.

__global__ __launch_bounds__(128)
void k_gemm_bf16(const unsigned short* __restrict__ A,
                 const unsigned short* __restrict__ Bp,
                 float* __restrict__ C, int M, int K,
                 const float* __restrict__ bias, int peRow, int accumulate) {
  const int lane = threadIdx.x & 31;
  const int wave = threadIdx.x >> 5;
  const int rowTiles = (M + 15) >> 4;
  const int totalTiles = rowTiles * 4;              // 4 col groups x 64 cols
  int tile = blockIdx.x * 4 + wave;
  if (tile >= totalTiles) return;
  const int rt = tile >> 2;
  const int cq = tile & 3;                          // column group (64 cols)
  const int half = lane >> 4;
  const int ln = lane & 15;
  const int m0 = rt * 16;
  const int kcn = K >> 5;

  v8f acc[4];
#pragma unroll
  for (int i = 0; i < 4; ++i) acc[i] = (v8f){0.f,0.f,0.f,0.f,0.f,0.f,0.f,0.f};

  // Clamp instead of predicating: an out-of-range lane computes values for a
  // row that the guarded store never writes, so no exec-mask dance needed.
  const int arow = min(m0 + ln, M - 1);
  const unsigned short* aptr = A + (size_t)arow * K + (half << 4);
  const unsigned short* bptr = Bp + ((((size_t)cq * 4) * kcn * 32 + lane) << 4);
  const int ctOff = kcn << 9;                       // elements between col tiles

  for (int kc = 0; kc < kcn; ++kc) {
    FragBF a;
    a.q[0] = *(const uint4*)(aptr);                 // K = base .. base+7
    a.q[1] = *(const uint4*)(aptr + 8);             // K = base+8 .. base+15
    aptr += 32;
#pragma unroll
    for (int i = 0; i < 4; ++i) {
      const unsigned short* bs = bptr + i * ctOff;  // unroll-constant offset
      FragBF b;
      b.q[0] = ((const uint4*)bs)[0];
      b.q[1] = ((const uint4*)bs)[1];
      acc[i] = __builtin_amdgcn_wmma_f32_16x16x32_bf16(false, a.v, false, b.v,
                                                       (short)0, acc[i], false, false);
    }
    bptr += 512;                                    // next 32-K chunk
  }

#pragma unroll
  for (int i = 0; i < 4; ++i) {
    const int col = cq * 64 + i * 16 + ln;
    float extra = 0.0f;
    if (bias) extra += bias[col];
    if (peRow >= 0) {                                // sinusoidal PE over D=256
      int ii = col & ~1;
      float div = expf(-logf(10000.0f) * (float)ii / 256.0f);
      float ang = (float)peRow * div;
      extra += (col & 1) ? cosf(ang) : sinf(ang);
    }
#pragma unroll
    for (int r = 0; r < 8; ++r) {
      int m = m0 + half * 8 + r;
      if (m < M) {
        size_t idx = (size_t)m * 256 + col;
        float v = acc[i][r] + extra;
        if (accumulate) v += C[idx];
        C[idx] = v;
      }
    }
  }
}

// ---- attention score kernels (one wave per item) ---------------------------

__global__ __launch_bounds__(256)
void k_score_hop1(const float* __restrict__ pos,
                  const int* __restrict__ esI, const int* __restrict__ edI,
                  const unsigned char* __restrict__ eattr,
                  const float* __restrict__ mid1, const float* __restrict__ dstp,
                  const float* __restrict__ Wb, const float* __restrict__ bb,
                  const float* __restrict__ Wu, const float* __restrict__ bu,
                  const float* __restrict__ attn,
                  float* __restrict__ score, unsigned int* __restrict__ gmax, int E) {
  __shared__ float sWu[40*256];
  __shared__ float sWb[2*256];
  __shared__ float sBias[256];
  __shared__ float sAttn[256];
  for (int i = threadIdx.x; i < 40*256; i += 256) sWu[i] = Wu[i];
  for (int i = threadIdx.x; i < 512;    i += 256) sWb[i] = Wb[i];
  { int i = threadIdx.x; sBias[i] = bb[i] + bu[i]; sAttn[i] = attn[i]; }
  __syncthreads();

  const int lane = threadIdx.x & 31;
  const int wid  = threadIdx.x >> 5;
  float localMax = -3.0e38f;

  for (int e = blockIdx.x * 8 + wid; e < E; e += gridDim.x * 8) {
    int s = esI[e], d = edI[e];
    float dx = pos[s*3+0]-pos[d*3+0];
    float dy = pos[s*3+1]-pos[d*3+1];
    float dz = pos[s*3+2]-pos[d*3+2];
    float d1 = sqrtf(dx*dx+dy*dy+dz*dz + 1e-12f);
    float d1c = fminf(fmaxf(d1, 0.05f), 10.0f);
    float ea = eattr[e] ? 1.0f : 0.0f;
    float nb = 1.0f - ea;
    float b0 = d1 * nb, b1 = d1 * d1 * nb;

    float rd[8] = {0,0,0,0,0,0,0,0};
    rbf_dot8(d1c, ea, sWu, lane, rd);

#pragma unroll
    for (int jj = 0; jj < 8; ++jj) {
      int c = jj*32 + lane;
      float dmat = b0*sWb[c] + b1*sWb[256+c] + sBias[c] + rd[jj];
      float av = (mid1[(size_t)s*256 + c] + dstp[(size_t)d*256 + c]) * dmat;
      av = (av > 0.0f) ? av : 0.2f * av;               // leaky_relu(0.2)
      float p = wave_sum(sAttn[c] * av);
      if (lane == 0) { score[(size_t)e*8 + jj] = p; localMax = fmaxf(localMax, p); }
    }
  }
  if (lane == 0 && localMax > -2.9e38f) atomicMax(gmax, ford(localMax));
}

__global__ __launch_bounds__(256)
void k_score_hop2(const float* __restrict__ pos,
                  const int* __restrict__ tsI, const int* __restrict__ tmI,
                  const int* __restrict__ tdI,
                  const float* __restrict__ mid2, const float* __restrict__ dstp,
                  const float* __restrict__ mid1,
                  const float* __restrict__ Wa, const float* __restrict__ ba,
                  const float* __restrict__ Wu, const float* __restrict__ bu,
                  const float* __restrict__ attn,
                  float* __restrict__ score, unsigned int* __restrict__ gmax, int T) {
  __shared__ float sWu[40*256];
  __shared__ float sWa[2*256];
  __shared__ float sBias[256];
  __shared__ float sAttn[256];
  for (int i = threadIdx.x; i < 40*256; i += 256) sWu[i] = Wu[i];
  for (int i = threadIdx.x; i < 512;    i += 256) sWa[i] = Wa[i];
  { int i = threadIdx.x; sBias[i] = ba[i] + bu[i]; sAttn[i] = attn[i]; }
  __syncthreads();

  const int lane = threadIdx.x & 31;
  const int wid  = threadIdx.x >> 5;
  float localMax = -3.0e38f;

  for (int e = blockIdx.x * 8 + wid; e < T; e += gridDim.x * 8) {
    int s = tsI[e], m = tmI[e], d = tdI[e];
    float ang = angle_at(pos, s, m, d);
    float d2 = distc(pos, s, d);

    float rd[8] = {0,0,0,0,0,0,0,0};
    rbf_dot8(d2, 1.0f, sWu, lane, rd);

#pragma unroll
    for (int jj = 0; jj < 8; ++jj) {
      int c = jj*32 + lane;
      float mult = ang*sWa[c] + ang*ang*sWa[256+c] + sBias[c] + rd[jj];
      float base = mid2[(size_t)s*256+c] + dstp[(size_t)d*256+c] + mid1[(size_t)m*256+c];
      float av = base * mult;
      av = (av > 0.0f) ? av : 0.2f * av;
      float p = wave_sum(sAttn[c] * av);
      if (lane == 0) { score[(size_t)e*8 + jj] = p; localMax = fmaxf(localMax, p); }
    }
  }
  if (lane == 0 && localMax > -2.9e38f) atomicMax(gmax, ford(localMax));
}

__global__ __launch_bounds__(256)
void k_score_hop3(const float* __restrict__ pos,
                  const int* __restrict__ qsI, const int* __restrict__ q2I,
                  const int* __restrict__ q1I, const int* __restrict__ qdI,
                  const float* __restrict__ srcp, const float* __restrict__ mid2,
                  const float* __restrict__ mid1, const float* __restrict__ dstp,
                  const float* __restrict__ Wd, const float* __restrict__ bd,
                  const float* __restrict__ Wu, const float* __restrict__ bu,
                  const float* __restrict__ attn,
                  float* __restrict__ score, unsigned int* __restrict__ gmax, int Q) {
  __shared__ float sWu[40*256];
  __shared__ float sWd[6*256];
  __shared__ float sBias[256];
  __shared__ float sAttn[256];
  for (int i = threadIdx.x; i < 40*256; i += 256) sWu[i] = Wu[i];
  for (int i = threadIdx.x; i < 6*256;  i += 256) sWd[i] = Wd[i];
  { int i = threadIdx.x; sBias[i] = bd[i] + bu[i]; sAttn[i] = attn[i]; }
  __syncthreads();

  const int lane = threadIdx.x & 31;
  const int wid  = threadIdx.x >> 5;
  float localMax = -3.0e38f;

  for (int e = blockIdx.x * 8 + wid; e < Q; e += gridDim.x * 8) {
    int i0 = qsI[e], i1 = q2I[e], i2 = q1I[e], i3 = qdI[e];
    float a1 = angle_at(pos, i0, i1, i2);
    float a2 = angle_at(pos, i1, i2, i3);
    float dh = dihedral4(pos, i0, i1, i2, i3);
    float d3 = distc(pos, i0, i3);

    float rd[8] = {0,0,0,0,0,0,0,0};
    rbf_dot8(d3, 1.0f, sWu, lane, rd);

#pragma unroll
    for (int jj = 0; jj < 8; ++jj) {
      int c = jj*32 + lane;
      float mult = a1*sWd[c] + a1*a1*sWd[256+c] + a2*sWd[512+c] + a2*a2*sWd[768+c]
                 + dh*sWd[1024+c] + dh*dh*sWd[1280+c] + sBias[c] + rd[jj];
      float base = srcp[(size_t)i0*256+c] + mid2[(size_t)i1*256+c]
                 + mid1[(size_t)i2*256+c] + dstp[(size_t)i3*256+c];
      float av = base * mult;
      av = (av > 0.0f) ? av : 0.2f * av;
      float p = wave_sum(sAttn[c] * av);
      if (lane == 0) { score[(size_t)e*8 + jj] = p; localMax = fmaxf(localMax, p); }
    }
  }
  if (lane == 0 && localMax > -2.9e38f) atomicMax(gmax, ford(localMax));
}

// ---- softmax passes --------------------------------------------------------

__global__ void k_expden(float* __restrict__ score, const int* __restrict__ dstIdx,
                         float* __restrict__ den, const unsigned int* __restrict__ gmax,
                         int M) {
  long long i = (long long)blockIdx.x * blockDim.x + threadIdx.x;
  if (i >= (long long)M * 8) return;
  float gm = ford_inv(*gmax);
  int e = (int)(i >> 3), h = (int)(i & 7);
  float w = expf(score[i] - gm);
  score[i] = w;
  atomicAdd(&den[(size_t)dstIdx[e]*8 + h], w);
}

__global__ __launch_bounds__(256)
void k_scatter(const float* __restrict__ msg, const int* __restrict__ srcIdx,
               const int* __restrict__ dstIdx, const float* __restrict__ score,
               const float* __restrict__ den, float* __restrict__ outv, int M) {
  int c = threadIdx.x;
  int h = c >> 5;
  int e = blockIdx.x;
  if (e >= M) return;
  int s = srcIdx[e], d = dstIdx[e];
  float w = score[(size_t)e*8 + h] / (den[(size_t)d*8 + h] + 1e-16f);
  atomicAdd(&outv[(size_t)d*256 + c], msg[(size_t)s*256 + c] * w);
}

// ---- fused LayerNorm + PReLU (one wave per node) ---------------------------

__global__ __launch_bounds__(256)
void k_ln_prelu(const float* __restrict__ xin, const float* __restrict__ gamma,
                const float* __restrict__ beta, const float* __restrict__ pw,
                float* __restrict__ out, int M) {
  const int lane = threadIdx.x & 31;
  const int wid  = threadIdx.x >> 5;
  float p = pw[0];
  int n = blockIdx.x * 8 + wid;
  if (n >= M) return;
  float v[8]; float s = 0.0f;
#pragma unroll
  for (int j = 0; j < 8; ++j) { v[j] = xin[(size_t)n*256 + j*32 + lane]; s += v[j]; }
  s = wave_sum(s);
  float mu = s * (1.0f / 256.0f);
  float q = 0.0f;
#pragma unroll
  for (int j = 0; j < 8; ++j) { float d = v[j] - mu; q += d * d; }
  q = wave_sum(q);
  float inv = rsqrtf(q * (1.0f / 256.0f) + 1e-5f);
#pragma unroll
  for (int j = 0; j < 8; ++j) {
    int c = j*32 + lane;
    float o = (v[j] - mu) * inv * gamma[c] + beta[c];
    out[(size_t)n*256 + c] = (o >= 0.0f) ? o : p * o;
  }
}

// ---------------------------------------------------------------------------

extern "C" void kernel_launch(void* const* d_in, const int* in_sizes, int n_in,
                              void* d_out, int out_size, void* d_ws, size_t ws_size,
                              hipStream_t stream) {
  (void)n_in; (void)out_size; (void)ws_size;

  const float* x            = (const float*)d_in[0];
  const float* pos          = (const float*)d_in[1];
  const int*   edge_index   = (const int*)d_in[2];
  const int*   triple_index = (const int*)d_in[3];
  const int*   quadra_index = (const int*)d_in[4];
  const unsigned char* edge_attr = (const unsigned char*)d_in[5];
  const float* W_src        = (const float*)d_in[6];
  const float* W_dst        = (const float*)d_in[7];
  const float* W_mid1       = (const float*)d_in[8];
  const float* W_mid2       = (const float*)d_in[9];
  const float* W_pos_bonded = (const float*)d_in[10];
  const float* b_pos_bonded = (const float*)d_in[11];
  const float* W_pos_unb    = (const float*)d_in[12];
  const float* b_pos_unb    = (const float*)d_in[13];
  const float* W_pos_unb1   = (const float*)d_in[14];
  const float* b_pos_unb1   = (const float*)d_in[15];
  const float* W_pos_unb2   = (const float*)d_in[16];
  const float* b_pos_unb2   = (const float*)d_in[17];
  const float* W_angle      = (const float*)d_in[18];
  const float* b_angle      = (const float*)d_in[19];
  const float* W_dihedral   = (const float*)d_in[20];
  const float* b_dihedral   = (const float*)d_in[21];
  const float* W_1hop       = (const float*)d_in[22];
  const float* b_1hop       = (const float*)d_in[23];
  const float* W_2hop       = (const float*)d_in[24];
  const float* b_2hop       = (const float*)d_in[25];
  const float* W_3hop       = (const float*)d_in[26];
  const float* b_3hop       = (const float*)d_in[27];
  const float* attn2        = (const float*)d_in[28];
  const float* attn3        = (const float*)d_in[29];
  const float* attn4        = (const float*)d_in[30];
  const float* W_res        = (const float*)d_in[31];
  const float* bias         = (const float*)d_in[32];
  const float* ln_gamma     = (const float*)d_in[33];
  const float* ln_beta      = (const float*)d_in[34];
  const float* prelu_w      = (const float*)d_in[35];

  const int Nn = in_sizes[0] / 128;   // 50000
  const int Ee = in_sizes[2] / 2;     // 400000
  const int Tt = in_sizes[3] / 3;     // 300000
  const int Qq = in_sizes[4] / 4;     // 200000

  // ---- workspace carving (256B aligned slices) ----
  char* base = (char*)d_ws;
  size_t off = 0;
  auto carve = [&](size_t bytes) -> void* {
    void* r = base + off;
    off += (bytes + 255) & ~(size_t)255;
    return r;
  };
  unsigned short* x_bf   = (unsigned short*)carve((size_t)Nn * 128 * 2);
  unsigned short* WpSrc  = (unsigned short*)carve(128 * 256 * 2);
  unsigned short* WpDst  = (unsigned short*)carve(128 * 256 * 2);
  unsigned short* WpMid1 = (unsigned short*)carve(128 * 256 * 2);
  unsigned short* WpMid2 = (unsigned short*)carve(128 * 256 * 2);
  unsigned short* WpRes  = (unsigned short*)carve(128 * 256 * 2);
  unsigned short* Wp1    = (unsigned short*)carve(256 * 256 * 2);
  unsigned short* Wp2    = (unsigned short*)carve(256 * 256 * 2);
  unsigned short* Wp3    = (unsigned short*)carve(256 * 256 * 2);
  float* projSrc  = (float*)carve((size_t)Nn * 256 * 4);
  float* projMid2 = (float*)carve((size_t)Nn * 256 * 4);
  float* projMid1 = (float*)carve((size_t)Nn * 256 * 4);
  float* projDst  = (float*)carve((size_t)Nn * 256 * 4);
  float* outAcc   = (float*)carve((size_t)Nn * 256 * 4);
  float* hopOut   = (float*)carve((size_t)Nn * 256 * 4);
  unsigned short* hopBf = (unsigned short*)carve((size_t)Nn * 256 * 2);
  float* score    = (float*)carve((size_t)Ee * 8 * 4);     // E >= T >= Q
  float* den      = (float*)carve((size_t)Nn * 8 * 4);
  unsigned int* gmax = (unsigned int*)carve(256);

  auto gemm = [&](const unsigned short* A, const unsigned short* Bp, float* C,
                  int M, int K, const float* b, int peRow, int accum) {
    int totalTiles = ((M + 15) >> 4) * 4;   // 16x64 strips
    int blocks = (totalTiles + 3) >> 2;
    k_gemm_bf16<<<blocks, 128, 0, stream>>>(A, Bp, C, M, K, b, peRow, accum);
  };
  auto pack = [&](const float* W, unsigned short* Bp, int K) {
    int total = K * 256;
    k_pack_w<<<(total + 255) / 256, 256, 0, stream>>>(W, Bp, K);
  };

  // ---- stage 0: bf16 conversion + weight repack ----
  {
    long long n = (long long)Nn * 128;
    k_f32_to_bf16<<<(unsigned)((n + 255) / 256), 256, 0, stream>>>(x, x_bf, n);
  }
  pack(W_src, WpSrc, 128);  pack(W_dst, WpDst, 128);
  pack(W_mid1, WpMid1, 128); pack(W_mid2, WpMid2, 128);
  pack(W_res, WpRes, 128);
  pack(W_1hop, Wp1, 256); pack(W_2hop, Wp2, 256); pack(W_3hop, Wp3, 256);

  // ---- stage 1: node projections (WMMA) with sinusoidal PE epilogue ----
  gemm(x_bf, WpSrc,  projSrc,  Nn, 128, nullptr, 0, 0);  // + pe[0]
  gemm(x_bf, WpMid2, projMid2, Nn, 128, nullptr, 1, 0);  // + pe[1]
  gemm(x_bf, WpMid1, projMid1, Nn, 128, nullptr, 2, 0);  // + pe[2]
  gemm(x_bf, WpDst,  projDst,  Nn, 128, nullptr, 3, 0);  // + pe[3]
  gemm(x_bf, WpRes,  outAcc,   Nn, 128, bias,   -1, 0);  // residual + bias

  const long long nOut = (long long)Nn * 256;
  const unsigned cvtBlocks = (unsigned)((nOut + 255) / 256);

  // ---- hop 1: edges ----
  hipMemsetAsync(hopOut, 0, (size_t)Nn * 256 * 4, stream);
  hipMemsetAsync(den, 0, (size_t)Nn * 8 * 4, stream);
  hipMemsetAsync(gmax, 0, 4, stream);
  k_score_hop1<<<2048, 256, 0, stream>>>(pos, edge_index, edge_index + Ee, edge_attr,
      projMid1, projDst, W_pos_bonded, b_pos_bonded, W_pos_unb, b_pos_unb,
      attn2, score, gmax, Ee);
  k_expden<<<(unsigned)(((long long)Ee * 8 + 255) / 256), 256, 0, stream>>>(
      score, edge_index + Ee, den, gmax, Ee);
  k_scatter<<<Ee, 256, 0, stream>>>(projMid1, edge_index, edge_index + Ee,
                                    score, den, hopOut, Ee);
  k_f32_to_bf16<<<cvtBlocks, 256, 0, stream>>>(hopOut, hopBf, nOut);
  gemm(hopBf, Wp1, outAcc, Nn, 256, b_1hop, -1, 1);

  // ---- hop 2: triples ----
  hipMemsetAsync(hopOut, 0, (size_t)Nn * 256 * 4, stream);
  hipMemsetAsync(den, 0, (size_t)Nn * 8 * 4, stream);
  hipMemsetAsync(gmax, 0, 4, stream);
  k_score_hop2<<<2048, 256, 0, stream>>>(pos, triple_index, triple_index + Tt,
      triple_index + 2 * Tt, projMid2, projDst, projMid1,
      W_angle, b_angle, W_pos_unb1, b_pos_unb1, attn3, score, gmax, Tt);
  k_expden<<<(unsigned)(((long long)Tt * 8 + 255) / 256), 256, 0, stream>>>(
      score, triple_index + 2 * Tt, den, gmax, Tt);
  k_scatter<<<Tt, 256, 0, stream>>>(projMid2, triple_index, triple_index + 2 * Tt,
                                    score, den, hopOut, Tt);
  k_f32_to_bf16<<<cvtBlocks, 256, 0, stream>>>(hopOut, hopBf, nOut);
  gemm(hopBf, Wp2, outAcc, Nn, 256, b_2hop, -1, 1);

  // ---- hop 3: quadruples ----
  hipMemsetAsync(hopOut, 0, (size_t)Nn * 256 * 4, stream);
  hipMemsetAsync(den, 0, (size_t)Nn * 8 * 4, stream);
  hipMemsetAsync(gmax, 0, 4, stream);
  k_score_hop3<<<2048, 256, 0, stream>>>(pos, quadra_index, quadra_index + Qq,
      quadra_index + 2 * Qq, quadra_index + 3 * Qq,
      projSrc, projMid2, projMid1, projDst,
      W_dihedral, b_dihedral, W_pos_unb2, b_pos_unb2, attn4, score, gmax, Qq);
  k_expden<<<(unsigned)(((long long)Qq * 8 + 255) / 256), 256, 0, stream>>>(
      score, quadra_index + 3 * Qq, den, gmax, Qq);
  k_scatter<<<Qq, 256, 0, stream>>>(projSrc, quadra_index, quadra_index + 3 * Qq,
                                    score, den, hopOut, Qq);
  k_f32_to_bf16<<<cvtBlocks, 256, 0, stream>>>(hopOut, hopBf, nOut);
  gemm(hopBf, Wp3, outAcc, Nn, 256, b_3hop, -1, 1);

  // ---- final: fused LayerNorm + PReLU -> d_out ----
  k_ln_prelu<<<(Nn + 7) / 8, 256, 0, stream>>>(outAcc, ln_gamma, ln_beta,
                                               prelu_w, (float*)d_out, Nn);
}
